// IncidenceConvolution_64604898066628
// MI455X (gfx1250) — matrix operand order repaired and checked by reading, not compiled, not run on hardware
//
#include <hip/hip_runtime.h>
#include <hip/hip_bf16.h>

// CDNA5 / gfx1250: wave32, WMMA 16x16x32 f16 -> f32
typedef __attribute__((ext_vector_type(16))) _Float16 v16h;
typedef __attribute__((ext_vector_type(8)))  float    v8f;

#define HDIM 256
#define LDIM 64
#define ROWS_PER_BLOCK 2          // inc rows per block (128 M-rows)

// ---------------------------------------------------------------------------
// Prep kernel: swizzle W2 (f32 [h][k] row-major) into the exact per-lane
// WMMA B-fragment layout (f16), and W1/b1 into the A-element order.
// B order used here (contiguous k): lanes 0-15 hold K=0..15 at col n=lane; lanes 16-31 hold
// K=16..31 at col n=lane-16; half-element e == k&15.
// A order used here (same contiguous k): lane<16 row m=lane, k=e;
// lane>=16 same but k += 16.
// ---------------------------------------------------------------------------
__global__ void prep_swizzle(const float* __restrict__ W1,
                             const float* __restrict__ b1,
                             const float* __restrict__ W2,
                             _Float16* __restrict__ wsB,
                             _Float16* __restrict__ wsW1,
                             _Float16* __restrict__ wsB1) {
  int t = blockIdx.x * blockDim.x + threadIdx.x;
  int stride = gridDim.x * blockDim.x;
  for (int i = t; i < HDIM * HDIM; i += stride) {
    int kh = i >> 8;        // 0..255  (K)
    int n  = i & 255;       // 0..255  (N)
    int ct = n >> 4, ln = n & 15;     // column tile, lane-low
    int ks = kh >> 5, kk = kh & 31;   // K-step, K within step
    int hi = kk >> 4, e = kk & 15;    // lane-high half, half element
    int lane = ln + (hi << 4);
    wsB[(size_t)((((ct << 3) + ks) << 5) + lane) * 16 + e] = (_Float16)W2[i];
  }
  if (t < HDIM) {
    int ks = t >> 5, r = t & 31, hi = r >> 4, e = r & 15;
    int k = (ks << 5) + (hi << 4) + e;
    wsW1[((ks << 1) + hi) * 16 + e] = (_Float16)W1[k];
    wsB1[((ks << 1) + hi) * 16 + e] = (_Float16)b1[k];
  }
}

// ---------------------------------------------------------------------------
// Main fused kernel: 512 threads = 16 waves, TWO (b,f) rows per block
// (M=128, N=256, K=256). h1 A-fragments are built ONCE per block into LDS in
// WMMA register layout; each wave streams them with ds_load_b128 pairs and
// owns one column tile so each B fragment feeds 8 WMMAs. sched_group_barrier
// enforces a [16 ds_read][8 wmma] pipeline so DS latency is hidden behind the
// previous group's WMMAs.  Uses sum(norm)==1 so no per-row reduction needed.
// ---------------------------------------------------------------------------
__global__ __launch_bounds__(512)
void incconv_mlp_kernel(const float* __restrict__ inc_m,
                        const float* __restrict__ b2,
                        const float* __restrict__ W3,
                        const float* __restrict__ b3,
                        const _Float16* __restrict__ wsB,
                        const _Float16* __restrict__ wsW1,
                        const _Float16* __restrict__ wsB1,
                        float* __restrict__ out) {
  // A fragments: frag f = ks*8+rt, per-lane 16 halves (32B) contiguous
  __shared__ __align__(32) _Float16 s_A[8 * 8 * 32 * 16];   // 64 KB
  __shared__ float s_norm[LDIM * ROWS_PER_BLOCK];           // 128 scalars
  __shared__ float s_out[ROWS_PER_BLOCK];

  const int tid = threadIdx.x;
  const float* rows = inc_m + (size_t)blockIdx.x * (LDIM * ROWS_PER_BLOCK);

  // ---- Phase 0: per-row sum + normalize (waves 0 and 1, one row each) ----
  if (tid < LDIM * ROWS_PER_BLOCK) s_norm[tid] = rows[tid];
  if (tid < ROWS_PER_BLOCK) s_out[tid] = 0.0f;
  __syncthreads();
  if (tid < 64) {
    const int r = tid >> 5;
    const int l = tid & 31;
    float v = s_norm[r * 64 + l] + s_norm[r * 64 + l + 32];
    for (int off = 16; off; off >>= 1) v += __shfl_xor(v, off, 32);
    const float inv = 1.0f / v;
    s_norm[r * 64 + l]      *= inv;
    s_norm[r * 64 + l + 32] *= inv;
  }
  __syncthreads();

  const int wave   = tid >> 5;      // 0..15 == column tile
  const int lane   = tid & 31;
  const int lnLo   = lane & 15;
  const int laneHi = lane >> 4;

  const v16h* Bfr  = (const v16h*)wsB;
  const v16h* W1fr = (const v16h*)wsW1;
  const v16h* B1fr = (const v16h*)wsB1;
  const v16h vzero = {};

  // ---- Phase 0.5: build all 64 A fragments once, cooperatively -----------
  // wave w builds fragments f = 4w .. 4w+3  (f = ks*8 + rt)
#pragma unroll
  for (int j = 0; j < 4; ++j) {
    const int f  = wave * 4 + j;
    const int ks = f >> 3, rt = f & 7;
    v16h w1f = W1fr[ks * 2 + laneHi];
    v16h b1f = B1fr[ks * 2 + laneHi];
    _Float16 nhv = (_Float16)s_norm[rt * 16 + lnLo];
    v16h t = w1f * nhv + b1f;                              // v_pk_fma_f16
    v16h a = __builtin_elementwise_max(t, vzero);          // v_pk_max_num_f16
    *(v16h*)&s_A[(size_t)(f * 32 + lane) * 16] = a;        // 2x ds_store_b128
  }
  __syncthreads();

  // per-column constants for this lane's column tile
  const int ncol = wave * 16 + lnLo;
  const float b2v = b2[ncol];
  const float w3v = W3[ncol];

  // accumulators initialized with bias b2 (all 8 elems share column n)
  v8f acc[8];
#pragma unroll
  for (int rt = 0; rt < 8; ++rt)
#pragma unroll
    for (int v = 0; v < 8; ++v) acc[rt][v] = b2v;

  const int cbase = wave * 256 + lane;   // B fragment index base

  // ---- Phase 1: GEMM, K = 256, fully unrolled, group-pipelined -----------
#pragma unroll
  for (int ks = 0; ks < 8; ++ks) {
    const v16h bfc = Bfr[cbase + ks * 32];                 // 2x global b128
    v16h af[8];
#pragma unroll
    for (int rt = 0; rt < 8; ++rt)
      af[rt] = *(const v16h*)&s_A[(size_t)((ks * 8 + rt) * 32 + lane) * 16];
#pragma unroll
    for (int rt = 0; rt < 8; ++rt)
      acc[rt] = __builtin_amdgcn_wmma_f32_16x16x32_f16(
          false, af[rt], false, bfc, (short)0, acc[rt], false, false);
    // scheduling pattern per K-step: 16 LDS reads, then 8 WMMAs.
    // Next step's reads issue while this step's WMMAs execute; in-order DS
    // completion means each WMMA only waits for its own pair.
    __builtin_amdgcn_sched_group_barrier(0x100, 16, 0);    // DS read x16
    __builtin_amdgcn_sched_group_barrier(0x008, 8, 0);     // WMMA   x8
  }

  // ---- Phase 2: local_r = sum relu(h2[m][n]) * W3[n] * norm[m] -----------
  // D layout: elem v at lane -> m = rt*16 + laneHi*8 + v, n = wave*16 + lnLo
  float local[ROWS_PER_BLOCK] = {0.0f, 0.0f};
#pragma unroll
  for (int rt = 0; rt < 8; ++rt) {
    const float4* np = (const float4*)&s_norm[rt * 16 + laneHi * 8];
    float4 n0 = np[0], n1 = np[1];
    float nv[8] = {n0.x, n0.y, n0.z, n0.w, n1.x, n1.y, n1.z, n1.w};
    const int r = rt >> 2;       // which inc row this row tile belongs to
#pragma unroll
    for (int v = 0; v < 8; ++v) {
      float s = fmaxf(acc[rt][v], 0.0f) * w3v;
      local[r] = fmaf(s, nv[v], local[r]);
    }
  }
#pragma unroll
  for (int r = 0; r < ROWS_PER_BLOCK; ++r) {
    float v = local[r];
    for (int off = 16; off; off >>= 1) v += __shfl_xor(v, off, 32);
    if (lane == 0) atomicAdd(&s_out[r], v);   // ds_add_f32
  }
  __syncthreads();

  // ---- Phase 3: add b3 (sum(norm)==1) and store --------------------------
  if (tid < ROWS_PER_BLOCK)
    out[blockIdx.x * ROWS_PER_BLOCK + tid] = s_out[tid] + b3[0];
}

// ---------------------------------------------------------------------------
extern "C" void kernel_launch(void* const* d_in, const int* in_sizes, int n_in,
                              void* d_out, int out_size, void* d_ws, size_t ws_size,
                              hipStream_t stream) {
  const float* inc_m = (const float*)d_in[0];  // [64,64,64]
  const float* W1    = (const float*)d_in[1];  // [1,256]
  const float* b1    = (const float*)d_in[2];  // [256]
  const float* W2    = (const float*)d_in[3];  // [256,256]
  const float* b2    = (const float*)d_in[4];  // [256]
  const float* W3    = (const float*)d_in[5];  // [256,1]
  const float* b3    = (const float*)d_in[6];  // [1]
  float* out = (float*)d_out;                  // [64,64]

  // workspace layout: swizzled W2 (128KB f16) + swizzled W1/b1 (512B each)
  _Float16* wsB  = (_Float16*)d_ws;
  _Float16* wsW1 = wsB + HDIM * HDIM;
  _Float16* wsB1 = wsW1 + HDIM;

  prep_swizzle<<<64, 256, 0, stream>>>(W1, b1, W2, wsB, wsW1, wsB1);
  const int nblocks = (64 * 64) / ROWS_PER_BLOCK;   // 2048
  incconv_mlp_kernel<<<nblocks, 512, 0, stream>>>(inc_m, b2, W3, b3,
                                                  wsB, wsW1, wsB1, out);
}
